// GE2ELoss_76536317214822
// MI455X (gfx1250) — compile-verified, matrix operand order checked by reading, not run on hardware
//
#include <hip/hip_runtime.h>
#include <stdint.h>

// GE2E loss, MI455X (gfx1250, wave32, WMMA + async-to-LDS staging).
// Sizes fixed by the reference: N=1024 speakers, M=20 utterances, D=768 dims.
#define N_SPK 1024
#define M_UTT 20
#define D_DIM 768
#define R_TOT (N_SPK * M_UTT)   // 20480 normalized embedding rows
#define GE2E_EPS 1e-8f

// GEMM tiling: workgroup tile 128(j) x 128(i), K-stage 32, double buffered.
// 8 waves arranged 4x2; wave tile 32(j) x 64(i) => 2 A-frags x 4 B-frags,
// 8 WMMAs per stage per wave with only 12 ds_load_b128.
#define TILE_J 128
#define TILE_I 128
// LDS row pitch in bf16 elements (32 K padded to 40 => 80B rows: 16 lanes of
// ds_load_b128 at 20-dword stride hit disjoint bank quads -> conflict-free).
#define LDS_PITCH 40

typedef __attribute__((ext_vector_type(16))) __bf16 v16bf;
typedef __attribute__((ext_vector_type(8)))  float  v8f;

union FragAB {
  v16bf v;
  uint4 q[2];   // two 16-byte chunks = 8 VGPRs of packed bf16
};

static __device__ __forceinline__ unsigned short f32_to_bf16_rne(float x) {
  union { float f; unsigned int u; } c; c.f = x;
  unsigned int u = c.u;
  u += 0x7FFFu + ((u >> 16) & 1u);   // round-to-nearest-even
  return (unsigned short)(u >> 16);
}

// ---- CDNA5 async global->LDS copy (ASYNCcnt-tracked), per-lane 16B ----------
static __device__ __forceinline__ void async_load_b128(unsigned int lds_byte_addr,
                                                       const void* gaddr) {
#if defined(__gfx1250__)
  asm volatile("global_load_async_to_lds_b128 %0, %1, off"
               :: "v"(lds_byte_addr), "v"(gaddr)
               : "memory");
#else
  (void)lds_byte_addr; (void)gaddr;
#endif
}
static __device__ __forceinline__ void wait_async_zero() {
#if defined(__gfx1250__)
  asm volatile("s_wait_asynccnt 0x0" ::: "memory");
#endif
}

static __device__ __forceinline__ float wave_reduce_sum(float v) {
  #pragma unroll
  for (int o = 16; o > 0; o >>= 1) v += __shfl_xor(v, o, 32);
  return v;
}
static __device__ __forceinline__ float wave_reduce_max(float v) {
  #pragma unroll
  for (int o = 16; o > 0; o >>= 1) v = fmaxf(v, __shfl_xor(v, o, 32));
  return v;
}

// Block-wide reductions for 256 threads (8 wave32s). Result broadcast to all.
static __device__ __forceinline__ float block_reduce_sum(float v) {
  __shared__ float s[8];
  v = wave_reduce_sum(v);
  if ((threadIdx.x & 31) == 0) s[threadIdx.x >> 5] = v;
  __syncthreads();
  if (threadIdx.x == 0) {
    float t = 0.f;
    #pragma unroll
    for (int i = 0; i < 8; ++i) t += s[i];
    s[0] = t;
  }
  __syncthreads();
  float r = s[0];
  __syncthreads();
  return r;
}
static __device__ __forceinline__ float block_reduce_max(float v) {
  __shared__ float s[8];
  v = wave_reduce_max(v);
  if ((threadIdx.x & 31) == 0) s[threadIdx.x >> 5] = v;
  __syncthreads();
  if (threadIdx.x == 0) {
    float t = s[0];
    #pragma unroll
    for (int i = 1; i < 8; ++i) t = fmaxf(t, s[i]);
    s[0] = t;
  }
  __syncthreads();
  float r = s[0];
  __syncthreads();
  return r;
}

// ---------------------------------------------------------------------------
// K1: L2-normalize each of the 20480 embedding vectors, emit bf16 row-major A.
// ---------------------------------------------------------------------------
__global__ __launch_bounds__(256) void ge2e_norm_rows(const float* __restrict__ E,
                                                      unsigned short* __restrict__ A) {
  const int r = blockIdx.x;
  const float* row = E + (size_t)r * D_DIM;
  float v0 = row[threadIdx.x];
  float v1 = row[threadIdx.x + 256];
  float v2 = row[threadIdx.x + 512];
  float ss = block_reduce_sum(v0 * v0 + v1 * v1 + v2 * v2);
  float scale = 1.0f / fmaxf(sqrtf(ss), GE2E_EPS);
  unsigned short* out = A + (size_t)r * D_DIM;
  out[threadIdx.x]       = f32_to_bf16_rne(v0 * scale);
  out[threadIdx.x + 256] = f32_to_bf16_rne(v1 * scale);
  out[threadIdx.x + 512] = f32_to_bf16_rne(v2 * scale);
}

// ---------------------------------------------------------------------------
// K2: per-speaker centroid (mean over M) + L2-normalize, emit bf16 row-major C.
// ---------------------------------------------------------------------------
__global__ __launch_bounds__(256) void ge2e_centroids(const float* __restrict__ E,
                                                      unsigned short* __restrict__ C) {
  const int j = blockIdx.x;
  const float* base = E + (size_t)j * M_UTT * D_DIM;
  float c[3];
  #pragma unroll
  for (int u = 0; u < 3; ++u) {
    const int d = threadIdx.x + u * 256;
    float s = 0.f;
    #pragma unroll 4
    for (int m = 0; m < M_UTT; ++m) s += base[(size_t)m * D_DIM + d];
    c[u] = s * (1.0f / (float)M_UTT);
  }
  float ss = block_reduce_sum(c[0] * c[0] + c[1] * c[1] + c[2] * c[2]);
  float scale = 1.0f / fmaxf(sqrtf(ss), GE2E_EPS);
  unsigned short* out = C + (size_t)j * D_DIM;
  #pragma unroll
  for (int u = 0; u < 3; ++u)
    out[threadIdx.x + u * 256] = f32_to_bf16_rne(c[u] * scale);
}

// ---------------------------------------------------------------------------
// K3: per utterance-slot m: S_m[j, i] = dot(c_j, e_{i,m}); stored so that i is
// contiguous:  G[(j*M + m)*N + i].
//
// Workgroup: 256 threads = 8 waves; tile 128 j x 128 i; wave tile 32 x 64.
// Double-buffered LDS K=32 slabs filled by global_load_async_to_lds_b128.
// All fragments for a stage are pulled into VGPRs first (clauseable ds_loads),
// then 8 back-to-back v_wmma_f32_16x16x32_bf16.
// Fragment VGPR layout (CDNA5, 16-bit A/B):
//   lanes 0..15  : K = 0..7 (q[0]) and 16..23 (q[1])
//   lanes 16..31 : K = 8..15 and 24..31, one matrix row/col per (lane & 15).
// ---------------------------------------------------------------------------
__global__ __launch_bounds__(256) void ge2e_gemm(const unsigned short* __restrict__ A,  // emb bf16 [20480 x 768]
                                                 const unsigned short* __restrict__ C,  // cent bf16 [1024 x 768]
                                                 float* __restrict__ G) {
  __shared__ unsigned short Asm[2][TILE_J * LDS_PITCH];   // 20,480 B
  __shared__ unsigned short Bsm[2][TILE_I * LDS_PITCH];   // 20,480 B

  const int lane = threadIdx.x & 31;
  const int wave = threadIdx.x >> 5;
  const int hl   = lane >> 4;        // half-wave select (K phase)
  const int ln   = lane & 15;

  const int j0 = blockIdx.x * TILE_J;   // centroid rows (output M dim)
  const int i0 = blockIdx.y * TILE_I;   // embedding rows (output N dim)
  const int m  = blockIdx.z;            // utterance slot

  const int jw = (wave & 3) * 32;       // wave's 32-row strip inside the tile
  const int ch = (wave >> 2) * 64;      // wave's 64-col strip inside the tile

  // Cooperative tile fill: 128 rows x 4 chunks(16B) = 512 chunks per tile,
  // 2 per thread per tile. id -> (row = id>>2, chunk = id&3).
  const int id0  = threadIdx.x;
  const int id1  = threadIdx.x + 256;
  const int rw0  = id0 >> 2, ck0 = (id0 & 3) * 8;     // chunk offset in bf16
  const int rw1  = id1 >> 2, ck1 = (id1 & 3) * 8;

  const unsigned short* gA0 = C + (size_t)(j0 + rw0) * D_DIM + ck0;
  const unsigned short* gA1 = C + (size_t)(j0 + rw1) * D_DIM + ck1;
  const unsigned short* gB0 = A + ((size_t)(i0 + rw0) * M_UTT + m) * D_DIM + ck0;
  const unsigned short* gB1 = A + ((size_t)(i0 + rw1) * M_UTT + m) * D_DIM + ck1;

  const unsigned int lA0 = (unsigned int)(uintptr_t)&Asm[0][rw0 * LDS_PITCH + ck0];
  const unsigned int lA1 = (unsigned int)(uintptr_t)&Asm[0][rw1 * LDS_PITCH + ck1];
  const unsigned int lB0 = (unsigned int)(uintptr_t)&Bsm[0][rw0 * LDS_PITCH + ck0];
  const unsigned int lB1 = (unsigned int)(uintptr_t)&Bsm[0][rw1 * LDS_PITCH + ck1];
  const unsigned int bufA = (unsigned int)(TILE_J * LDS_PITCH * 2);  // bytes
  const unsigned int bufB = (unsigned int)(TILE_I * LDS_PITCH * 2);

  v8f acc[8];
  #pragma unroll
  for (int t = 0; t < 8; ++t)
    #pragma unroll
    for (int e = 0; e < 8; ++e) acc[t][e] = 0.0f;

  // ---- stage kk = 0 into buffer 0 ----
  async_load_b128(lA0, gA0);
  async_load_b128(lA1, gA1);
  async_load_b128(lB0, gB0);
  async_load_b128(lB1, gB1);
  wait_async_zero();
  __syncthreads();

  int buf = 0;
  for (int kk = 0; kk < D_DIM; kk += 32) {
    const int nbuf = buf ^ 1;
    if (kk + 32 < D_DIM) {   // prefetch next K-slab into the other buffer
      const unsigned int bo = (unsigned int)nbuf;
      async_load_b128(lA0 + bo * bufA, gA0 + kk + 32);
      async_load_b128(lA1 + bo * bufA, gA1 + kk + 32);
      async_load_b128(lB0 + bo * bufB, gB0 + kk + 32);
      async_load_b128(lB1 + bo * bufB, gB1 + kk + 32);
    }

    // Pull ALL fragments for this stage into VGPRs, then the 8 WMMAs.
    FragAB a0, a1, bq[4];
    {
      const unsigned short* ar0 = &Asm[buf][(jw + ln) * LDS_PITCH];
      const unsigned short* ar1 = &Asm[buf][(jw + 16 + ln) * LDS_PITCH];
      a0.q[0] = *(const uint4*)(ar0 + hl * 8);
      a0.q[1] = *(const uint4*)(ar0 + 16 + hl * 8);
      a1.q[0] = *(const uint4*)(ar1 + hl * 8);
      a1.q[1] = *(const uint4*)(ar1 + 16 + hl * 8);
      #pragma unroll
      for (int t = 0; t < 4; ++t) {
        const unsigned short* br = &Bsm[buf][(ch + t * 16 + ln) * LDS_PITCH];
        bq[t].q[0] = *(const uint4*)(br + hl * 8);
        bq[t].q[1] = *(const uint4*)(br + 16 + hl * 8);
      }
    }
    #pragma unroll
    for (int t = 0; t < 4; ++t)
      acc[t] = __builtin_amdgcn_wmma_f32_16x16x32_bf16(
          false, a0.v, false, bq[t].v, (short)0, acc[t], false, false);
    #pragma unroll
    for (int t = 0; t < 4; ++t)
      acc[4 + t] = __builtin_amdgcn_wmma_f32_16x16x32_bf16(
          false, a1.v, false, bq[t].v, (short)0, acc[4 + t], false, false);

    wait_async_zero();   // own async fills done before the group barrier
    __syncthreads();
    buf = nbuf;
  }

  // C/D layout: VGPR g -> output row (g + 8*hl), column = lane & 15.
  // Store S_m[j, i] at G[(j*M + m)*N + i]  (i contiguous for the loss kernel).
  #pragma unroll
  for (int rt = 0; rt < 2; ++rt) {
    const int jbase = j0 + jw + rt * 16 + hl * 8;
    #pragma unroll
    for (int t = 0; t < 4; ++t) {
      const int i = i0 + ch + t * 16 + ln;
      #pragma unroll
      for (int g = 0; g < 8; ++g) {
        const int j = jbase + g;
        G[((size_t)j * M_UTT + m) * N_SPK + i] = acc[rt * 4 + t][g];
      }
    }
  }
}

// ---------------------------------------------------------------------------
// K4: logits row r = m*N + j lives at G + (j*M + m)*N (contiguous in i).
// partial[r] = (w*G_row[i*] + b) - logsumexp_i(w*G_row[i] + b),  i* = r / M.
// ---------------------------------------------------------------------------
__global__ __launch_bounds__(256) void ge2e_row_loss(const float* __restrict__ G,
                                                     const float* __restrict__ wp,
                                                     const float* __restrict__ bp,
                                                     float* __restrict__ partials) {
  const int r = blockIdx.x;          // r = m*N + j
  const int j = r & (N_SPK - 1);
  const int m = r >> 10;
  const float w = wp[0], b = bp[0];
  const float* row = G + ((size_t)j * M_UTT + m) * N_SPK;

  float vals[4];
  #pragma unroll
  for (int u = 0; u < 4; ++u)
    vals[u] = w * row[threadIdx.x + u * 256] + b;

  float mx = fmaxf(fmaxf(vals[0], vals[1]), fmaxf(vals[2], vals[3]));
  mx = block_reduce_max(mx);
  float se = expf(vals[0] - mx) + expf(vals[1] - mx) +
             expf(vals[2] - mx) + expf(vals[3] - mx);
  se = block_reduce_sum(se);
  if (threadIdx.x == 0) {
    const int lbl = r / M_UTT;       // target column i*
    const float tv = w * row[lbl] + b;
    partials[r] = tv - (mx + logf(se));
  }
}

// ---------------------------------------------------------------------------
// K5: loss = -mean(partials). Single block, fixed reduction tree.
// ---------------------------------------------------------------------------
__global__ __launch_bounds__(256) void ge2e_finalize(const float* __restrict__ partials,
                                                     float* __restrict__ out) {
  float s = 0.f;
  for (int k = threadIdx.x; k < R_TOT; k += 256) s += partials[k];
  s = block_reduce_sum(s);
  if (threadIdx.x == 0) out[0] = -s * (1.0f / (float)R_TOT);
}

extern "C" void kernel_launch(void* const* d_in, const int* in_sizes, int n_in,
                              void* d_out, int out_size, void* d_ws, size_t ws_size,
                              hipStream_t stream) {
  (void)in_sizes; (void)n_in; (void)out_size; (void)ws_size;
  const float* E  = (const float*)d_in[0];   // [1024, 20, 768] fp32
  const float* wI = (const float*)d_in[1];   // scalar
  const float* bI = (const float*)d_in[2];   // scalar
  float* out = (float*)d_out;

  // Workspace layout (all 16B-aligned offsets):
  //   A bf16 [20480 x 768]   : 31,457,280 B   (normalized embeddings)
  //   C bf16 [ 1024 x 768]   :  1,572,864 B   (normalized centroids)
  //   G f32  [1024*20 x 1024]: 83,886,080 B   (sims, row = (j,m), col = i)
  //   partials f32 [20480]   :     81,920 B
  char* ws = (char*)d_ws;
  unsigned short* Abf = (unsigned short*)ws;
  unsigned short* Cbf = (unsigned short*)(ws + (size_t)31457280);
  float* G            = (float*)(ws + (size_t)31457280 + 1572864);
  float* partials     = (float*)(ws + (size_t)31457280 + 1572864 + 83886080);

  ge2e_norm_rows<<<R_TOT, 256, 0, stream>>>(E, Abf);
  ge2e_centroids<<<N_SPK, 256, 0, stream>>>(E, Cbf);

  dim3 grid(N_SPK / TILE_J, N_SPK / TILE_I, M_UTT);   // 8 x 8 x 20 workgroups
  ge2e_gemm<<<grid, 256, 0, stream>>>(Abf, Cbf, G);

  ge2e_row_loss<<<R_TOT, 256, 0, stream>>>(G, wI, bI, partials);
  ge2e_finalize<<<1, 256, 0, stream>>>(partials, out);
}
